// transformer_19018115187307
// MI455X (gfx1250) — compile-verified, matrix operand order checked by reading, not run on hardware
//
#include <hip/hip_runtime.h>

#define HH 128
#define WW 512
#define ENC 32
#define BB 2
#define HW (HH * WW)        // 65536
#define CHW (ENC * HW)      // 2097152 = 2^21
#define NTOT (BB * CHW)     // 4194304
#define INV_SCALER 0.17677669529663689f  // 1/sqrt(32)

typedef __attribute__((ext_vector_type(16))) __bf16          v16bf;
typedef __attribute__((ext_vector_type(16))) unsigned short  v16u;
typedef __attribute__((ext_vector_type(8)))  unsigned short  v8u;
typedef __attribute__((ext_vector_type(8)))  float           v8f;

// ---------- helpers ----------

__device__ inline unsigned short f2bf(float f) {
  unsigned u = __float_as_uint(f);
  return (unsigned short)((u + 0x7FFFu + ((u >> 16) & 1u)) >> 16);  // RNE
}

__device__ inline v8f wmma_bf16(v16u a, v16u b, v8f c) {
  return __builtin_amdgcn_wmma_f32_16x16x32_bf16(
      false, __builtin_bit_cast(v16bf, a),
      false, __builtin_bit_cast(v16bf, b),
      (short)0, c, false, false);
}

// K-index map for 16-bit 16x32 A/B fragments (CDNA5 ISA 7.12.2):
// lanes 0-15 hold K {0..7, 16..23}; lanes 16-31 hold K {8..15, 24..31}
__device__ inline int kmap(int i, int half) { return (i < 8 ? i : i + 8) + half * 8; }

// K-contiguous fragment: two b128 loads per lane (requires 16B-aligned rows)
__device__ inline v16u frag_row(const unsigned short* p, int lda) {
  int lane = threadIdx.x & 31, half = lane >> 4, r = lane & 15;
  const unsigned short* q = p + r * lda + half * 8;
  v8u lo = *(const v8u*)(q);        // K = half*8 .. half*8+7
  v8u hi = *(const v8u*)(q + 16);   // K = 16+half*8 .. +7
  return __builtin_shufflevector(lo, hi, 0, 1, 2, 3, 4, 5, 6, 7,
                                 8, 9, 10, 11, 12, 13, 14, 15);
}

// ---------- f32 -> bf16 tensor conversion ----------
__global__ void tobf_k(const float* __restrict__ a, unsigned short* __restrict__ o, int n) {
  int i = blockIdx.x * 256 + threadIdx.x;
  if (i < n) o[i] = f2bf(a[i]);
}

// ---------- per-plane bf16 transpose: [HH x WW] -> [WW x HH] ----------
// grid: BB*ENC*64 blocks (32x32 tiles) of 256 threads
__global__ void tr_k(const unsigned short* __restrict__ src,
                     unsigned short* __restrict__ dst) {
  __shared__ unsigned short tile[32][33];
  int blk   = blockIdx.x;
  int plane = blk >> 6;          // 4 x 16 tiles per plane
  int trem  = blk & 63;
  int ty = (trem >> 4) * 32;     // h base
  int tx = (trem & 15) * 32;     // w base
  const unsigned short* s = src + plane * HW;
  unsigned short*       d = dst + plane * HW;
  int t = threadIdx.x;
  int r = t >> 5, c = t & 31;    // 8 rows per pass
  for (int rr = r; rr < 32; rr += 8)
    tile[rr][c] = s[(ty + rr) * WW + tx + c];
  __syncthreads();
  for (int rr = r; rr < 32; rr += 8)
    d[(tx + rr) * HH + ty + c] = tile[c][rr];
}

// ---------- conv weight packing: f32 [OC,IC,KS,KS] -> bf16 [tap][32][32] ----------
__global__ void pack_w_k(const float* __restrict__ wgt, int IC, int KS,
                         unsigned short* __restrict__ out) {
  int i = blockIdx.x * 256 + threadIdx.x;
  int total = KS * KS * ENC * 32;
  if (i >= total) return;
  int ic = i & 31, oc = (i >> 5) & 31, tap = i >> 10;
  int ky = tap / KS, kx = tap % KS;
  float w = (ic < IC) ? wgt[((oc * IC + ic) * KS + ky) * KS + kx] : 0.f;
  out[(tap * ENC + oc) * 32 + ic] = f2bf(w);
}

// ---------- implicit-GEMM conv (OC=32; KS=1 or 3), bf16 WMMA, f32 accum ----------
// grid: BB*HH*(WW/16) blocks of 64 threads (2 waves; wave w -> oc tile w)
__global__ void conv_wmma_k(const unsigned short* __restrict__ in, int IC,
                            const unsigned short* __restrict__ wp,  // packed bf16
                            const float* __restrict__ bias,
                            int KS, int pad, void* __restrict__ out,
                            int out_bf16, int relu) {
  int blk = blockIdx.x;
  int b   = blk >> 12;               // 4096 tiles per batch
  int rem = blk & 4095;
  int y   = rem >> 5;
  int x0  = (rem & 31) << 4;
  int tid = threadIdx.x;
  int wave = tid >> 5, lane = tid & 31, half = lane >> 4, r = lane & 15;
  int oc0 = wave * 16;

  v8f acc;
#pragma unroll
  for (int i = 0; i < 8; ++i) acc[i] = 0.f;

  for (int ky = 0; ky < KS; ++ky) {
    int yy = y + ky - pad;
    bool yok = (yy >= 0) && (yy < HH);
    for (int kx = 0; kx < KS; ++kx) {
      int tap = ky * KS + kx;
      int xx = x0 + r + kx - pad;
      bool ok = yok && (xx >= 0) && (xx < WW);
      v16u aw = frag_row(wp + (tap * ENC + oc0) * 32, 32);
      v16u bv;
      const unsigned short* ip = in + (b * IC) * HW + yy * WW + xx;
#pragma unroll
      for (int i = 0; i < 16; ++i) {
        int ic = kmap(i, half);
        bv[i] = (ok && ic < IC) ? ip[ic * HW] : (unsigned short)0;
      }
      acc = wmma_bf16(aw, bv, acc);   // chained accumulation across taps
    }
  }
#pragma unroll
  for (int v = 0; v < 8; ++v) {
    int oc = oc0 + v + half * 8;
    float f = acc[v] + bias[oc];
    if (relu) f = fmaxf(f, 0.f);
    int idx = ((b * ENC + oc) * HH + y) * WW + x0 + r;
    if (out_bf16) ((unsigned short*)out)[idx] = f2bf(f);
    else          ((float*)out)[idx] = f;
  }
}

// ---------- t1 = softmax_row(Q K^T / s); A = t1 @ V ----------
// grid: BB*ENC*4 blocks (32-row M blocks) of 256 threads (8 waves)
__global__ void attn_t1a_k(const unsigned short* __restrict__ Q,
                           const unsigned short* __restrict__ K,
                           const unsigned short* __restrict__ Vt,  // [w][g] transposed
                           unsigned short* __restrict__ A) {
  __shared__ float                      s1[32 * 128];   // 16 KB
  __shared__ alignas(16) unsigned short t1[32 * 128];   //  8 KB
  int bc    = blockIdx.x >> 2;
  int mbase = (blockIdx.x & 3) * 32;
  int plane = bc * HW;
  int tid = threadIdx.x, wave = tid >> 5, lane = tid & 31, half = lane >> 4, r = lane & 15;

  // phase 1: S1 block [32 x 128], K = 512 (16 WMMA / tile); both frags contiguous
  for (int t = wave; t < 16; t += 8) {
    int mt = t >> 3, nt = t & 7;
    v8f acc;
#pragma unroll
    for (int i = 0; i < 8; ++i) acc[i] = 0.f;
    const unsigned short* qp = Q + plane + (mbase + mt * 16) * WW;
    const unsigned short* kp = K + plane + (nt * 16) * WW;   // B frag = rows of K (K^T)
    for (int kk = 0; kk < WW; kk += 32) {
      __builtin_prefetch(qp + kk + 128, 0, 1);
      __builtin_prefetch(kp + kk + 128, 0, 1);
      acc = wmma_bf16(frag_row(qp + kk, WW), frag_row(kp + kk, WW), acc);
    }
#pragma unroll
    for (int v = 0; v < 8; ++v)
      s1[(mt * 16 + v + half * 8) * 128 + nt * 16 + r] = acc[v] * INV_SCALER;
  }
  __syncthreads();

  // phase 2: row softmax (axis=-1), emit bf16 t1
  if (tid < 32) {
    float* row = s1 + tid * 128;
    float m = -1e30f;
    for (int g = 0; g < 128; ++g) m = fmaxf(m, row[g]);
    float s = 0.f;
    for (int g = 0; g < 128; ++g) { float e = __expf(row[g] - m); row[g] = e; s += e; }
    float inv = 1.f / s;
    for (int g = 0; g < 128; ++g) t1[tid * 128 + g] = f2bf(row[g] * inv);
  }
  __syncthreads();

  // phase 3: A block [32 x 512], K = 128 ; B frag = rows of Vt (all contiguous)
  for (int t = wave; t < 64; t += 8) {
    int mt = t >> 5, nt = t & 31;
    v8f acc;
#pragma unroll
    for (int i = 0; i < 8; ++i) acc[i] = 0.f;
    for (int kk = 0; kk < 128; kk += 32)
      acc = wmma_bf16(frag_row(t1 + mt * 16 * 128 + kk, 128),
                      frag_row(Vt + plane + (nt * 16) * HH + kk, HH), acc);
#pragma unroll
    for (int v = 0; v < 8; ++v)
      A[plane + (mbase + mt * 16 + v + half * 8) * WW + nt * 16 + r] = f2bf(acc[v]);
  }
}

// ---------- t2 column block = softmax_col(Q^T K / s); Z = A @ t2 ----------
// grid: BB*ENC*32 blocks (16-column N tiles) of 256 threads; t2 never hits HBM
__global__ void attn_t2o_k(const unsigned short* __restrict__ Qt,  // [w][h]
                           const unsigned short* __restrict__ Kt,  // [v][h]
                           const unsigned short* __restrict__ A,
                           float* __restrict__ Z) {
  __shared__ float                      s2[512 * 16];    // 32 KB
  __shared__ alignas(16) unsigned short t2t[16 * 512];   // 16 KB, [v][w]
  __shared__ float                      red[16 * 16];
  __shared__ float                      colv[16];
  int bc    = blockIdx.x >> 5;
  int vbase = (blockIdx.x & 31) * 16;
  int plane = bc * HW;
  int tid = threadIdx.x, wave = tid >> 5, lane = tid & 31, half = lane >> 4, r = lane & 15;

  // phase 1: S2[w, v] = sum_h Q[h,w] K[h,v] / s  (32 w-tiles, K = 128, all contiguous)
  for (int wt = wave; wt < 32; wt += 8) {
    int wbase = wt * 16;
    v8f acc;
#pragma unroll
    for (int i = 0; i < 8; ++i) acc[i] = 0.f;
    const unsigned short* qp = Qt + plane + wbase * HH;   // rows = columns of Q
    const unsigned short* kp = Kt + plane + vbase * HH;   // rows = columns of K
    for (int kk = 0; kk < 128; kk += 32)
      acc = wmma_bf16(frag_row(qp + kk, HH), frag_row(kp + kk, HH), acc);
#pragma unroll
    for (int v = 0; v < 8; ++v)
      s2[(wbase + v + half * 8) * 16 + r] = acc[v] * INV_SCALER;
  }
  __syncthreads();

  // phase 2: softmax over w (axis=-2), per column; emit transposed bf16 t2t[v][w]
  int c = tid & 15, grp = tid >> 4;
  float m = -1e30f;
  for (int j = 0; j < 32; ++j) m = fmaxf(m, s2[(grp * 32 + j) * 16 + c]);
  red[grp * 16 + c] = m;
  __syncthreads();
  if (tid < 16) {
    float mm = -1e30f;
    for (int j = 0; j < 16; ++j) mm = fmaxf(mm, red[j * 16 + tid]);
    colv[tid] = mm;
  }
  __syncthreads();
  float cm = colv[c];
  float s = 0.f;
  for (int j = 0; j < 32; ++j) {
    int w = grp * 32 + j;
    float e = __expf(s2[w * 16 + c] - cm);
    s2[w * 16 + c] = e; s += e;
  }
  __syncthreads();
  red[grp * 16 + c] = s;
  __syncthreads();
  if (tid < 16) {
    float ss = 0.f;
    for (int j = 0; j < 16; ++j) ss += red[j * 16 + tid];
    colv[tid] = ss;
  }
  __syncthreads();
  float inv = 1.f / colv[c];
  for (int j = 0; j < 32; ++j) {
    int w = grp * 32 + j;
    t2t[c * 512 + w] = f2bf(s2[w * 16 + c] * inv);
  }
  __syncthreads();

  // phase 3: Z[h, vbase..+16) = A[h,:] @ t2  (8 h-tiles, K = 512; b128 ds loads)
  {
    int mt = wave;
    v8f acc;
#pragma unroll
    for (int i = 0; i < 8; ++i) acc[i] = 0.f;
    const unsigned short* ap = A + plane + mt * 16 * WW;
    for (int kk = 0; kk < WW; kk += 32) {
      __builtin_prefetch(ap + kk + 128, 0, 1);
      acc = wmma_bf16(frag_row(ap + kk, WW), frag_row(t2t + kk, 512), acc);
    }
#pragma unroll
    for (int v = 0; v < 8; ++v)
      Z[plane + (mt * 16 + v + half * 8) * WW + vbase + r] = acc[v];
  }
}

// ---------- LayerNorm over (C,H,W), fused residual add, deterministic ----------
__global__ void ln_reduce_k(const float* __restrict__ a, const float* __restrict__ b,
                            float* __restrict__ part) {
  __shared__ float ssum[256], ssq[256];
  int sidx = blockIdx.x >> 8;   // 256 partials per sample
  int p    = blockIdx.x & 255;
  size_t base = (size_t)sidx * CHW + (size_t)p * (CHW / 256);
  float sum = 0.f, sq = 0.f;
  for (int j = threadIdx.x; j < CHW / 256; j += 256) {
    float v = a[base + j] + b[base + j];
    sum += v; sq += v * v;
  }
  ssum[threadIdx.x] = sum; ssq[threadIdx.x] = sq;
  __syncthreads();
  for (int off = 128; off > 0; off >>= 1) {
    if (threadIdx.x < off) {
      ssum[threadIdx.x] += ssum[threadIdx.x + off];
      ssq[threadIdx.x]  += ssq[threadIdx.x + off];
    }
    __syncthreads();
  }
  if (threadIdx.x == 0) {
    part[(sidx * 256 + p) * 2 + 0] = ssum[0];
    part[(sidx * 256 + p) * 2 + 1] = ssq[0];
  }
}

__global__ void ln_stats_k(const float* __restrict__ part, float* __restrict__ stats) {
  __shared__ float ssum[256], ssq[256];
  int sidx = blockIdx.x;
  ssum[threadIdx.x] = part[(sidx * 256 + threadIdx.x) * 2 + 0];
  ssq[threadIdx.x]  = part[(sidx * 256 + threadIdx.x) * 2 + 1];
  __syncthreads();
  for (int off = 128; off > 0; off >>= 1) {
    if (threadIdx.x < off) {
      ssum[threadIdx.x] += ssum[threadIdx.x + off];
      ssq[threadIdx.x]  += ssq[threadIdx.x + off];
    }
    __syncthreads();
  }
  if (threadIdx.x == 0) {
    float mean = ssum[0] / (float)CHW;
    float var  = ssq[0] / (float)CHW - mean * mean;
    stats[sidx * 2 + 0] = mean;
    stats[sidx * 2 + 1] = rsqrtf(var + 1e-5f);
  }
}

__global__ void ln_apply_k(const float* __restrict__ a, const float* __restrict__ b,
                           const float* __restrict__ g, const float* __restrict__ bt,
                           const float* __restrict__ stats, float* __restrict__ out) {
  for (size_t i = (size_t)blockIdx.x * 256 + threadIdx.x; i < (size_t)NTOT;
       i += (size_t)gridDim.x * 256) {
    int sidx = (int)(i >> 21);            // CHW = 2^21
    int j    = (int)(i & (CHW - 1));
    float v = a[i] + b[i];
    out[i] = (v - stats[sidx * 2]) * stats[sidx * 2 + 1] * g[j] + bt[j];
  }
}

// ---------- final 1x1 conv ENC -> 1 ----------
__global__ void final_conv_k(const float* __restrict__ X, const float* __restrict__ w,
                             const float* __restrict__ bias, float* __restrict__ out) {
  int i = blockIdx.x * 256 + threadIdx.x;   // BB*HW = 131072
  if (i >= BB * HW) return;
  int b = i >> 16, hw = i & (HW - 1);
  float s = bias[0];
  for (int cc = 0; cc < ENC; ++cc) s += w[cc] * X[(b * ENC + cc) * HW + hw];
  out[i] = s;
}

// ---------- host ----------
extern "C" void kernel_launch(void* const* d_in, const int* in_sizes, int n_in,
                              void* d_out, int out_size, void* d_ws, size_t ws_size,
                              hipStream_t stream) {
  (void)in_sizes; (void)n_in; (void)out_size; (void)ws_size;
  const size_t F32B = (size_t)NTOT * 4;   // 16.78 MB
  const size_t B16B = (size_t)NTOT * 2;   //  8.39 MB
  char* ws = (char*)d_ws;
  float*          X   = (float*)(ws);
  float*          Z   = (float*)(ws + F32B);
  float*          RES = (float*)(ws + 2 * F32B);   // also reused as Z2
  float*          ZN  = (float*)(ws + 3 * F32B);
  unsigned short* Qb  = (unsigned short*)(ws + 4 * F32B);
  unsigned short* Kb  = (unsigned short*)(ws + 4 * F32B + 1 * B16B);
  unsigned short* Vb  = (unsigned short*)(ws + 4 * F32B + 2 * B16B);
  unsigned short* Ab  = (unsigned short*)(ws + 4 * F32B + 3 * B16B);
  unsigned short* Xb  = (unsigned short*)(ws + 4 * F32B + 4 * B16B);  // bf16 conv input
  unsigned short* Qt  = (unsigned short*)(ws + 4 * F32B + 5 * B16B);  // transposed
  unsigned short* Kt  = (unsigned short*)(ws + 4 * F32B + 6 * B16B);
  unsigned short* Vt  = (unsigned short*)(ws + 4 * F32B + 7 * B16B);
  unsigned short* WP  = (unsigned short*)(ws + 4 * F32B + 8 * B16B);  // packed weights
  float*          PART  = (float*)(ws + 4 * F32B + 8 * B16B + 32768);
  float*          STATS = PART + 2048;

  const float* x_in = (const float*)d_in[0];
  for (int L = 0; L < 4; ++L) {
    const float* in_f = (L == 0) ? x_in : X;
    int IC = (L == 0) ? 8 : ENC;
    int n_in_elems = BB * IC * HW;
    int base = 2 + 14 * L;
    const float* wq_w = (const float*)d_in[base + 0];
    const float* wq_b = (const float*)d_in[base + 1];
    const float* wk_w = (const float*)d_in[base + 2];
    const float* wk_b = (const float*)d_in[base + 3];
    const float* wv_w = (const float*)d_in[base + 4];
    const float* wv_b = (const float*)d_in[base + 5];
    const float* rs_w = (const float*)d_in[base + 6];
    const float* rs_b = (const float*)d_in[base + 7];
    const float* ln_w = (const float*)d_in[base + 8];
    const float* ln_b = (const float*)d_in[base + 9];
    const float* n1_w = (const float*)d_in[base + 10];
    const float* n1_b = (const float*)d_in[base + 11];
    const float* n2_w = (const float*)d_in[base + 12];
    const float* n2_b = (const float*)d_in[base + 13];

    tobf_k<<<(n_in_elems + 255) / 256, 256, 0, stream>>>(in_f, Xb, n_in_elems);

    pack_w_k<<<4, 256, 0, stream>>>(wq_w, IC, 1, WP);
    conv_wmma_k<<<8192, 64, 0, stream>>>(Xb, IC, WP, wq_b, 1, 0, Qb, 1, 0);
    pack_w_k<<<36, 256, 0, stream>>>(wk_w, IC, 3, WP);
    conv_wmma_k<<<8192, 64, 0, stream>>>(Xb, IC, WP, wk_b, 3, 1, Kb, 1, 0);
    pack_w_k<<<36, 256, 0, stream>>>(wv_w, IC, 3, WP);
    conv_wmma_k<<<8192, 64, 0, stream>>>(Xb, IC, WP, wv_b, 3, 1, Vb, 1, 0);
    pack_w_k<<<36, 256, 0, stream>>>(rs_w, IC, 3, WP);
    conv_wmma_k<<<8192, 64, 0, stream>>>(Xb, IC, WP, rs_b, 3, 1, RES, 0, 0);

    tr_k<<<BB * ENC * 64, 256, 0, stream>>>(Qb, Qt);
    tr_k<<<BB * ENC * 64, 256, 0, stream>>>(Kb, Kt);
    tr_k<<<BB * ENC * 64, 256, 0, stream>>>(Vb, Vt);

    attn_t1a_k<<<BB * ENC * 4, 256, 0, stream>>>(Qb, Kb, Vt, Ab);
    attn_t2o_k<<<BB * ENC * 32, 256, 0, stream>>>(Qt, Kt, Ab, Z);

    ln_reduce_k<<<BB * 256, 256, 0, stream>>>(Z, RES, PART);
    ln_stats_k<<<BB, 256, 0, stream>>>(PART, STATS);
    ln_apply_k<<<2048, 256, 0, stream>>>(Z, RES, n1_w, n1_b, STATS, ZN);

    tobf_k<<<(NTOT + 255) / 256, 256, 0, stream>>>(ZN, Xb, NTOT);
    pack_w_k<<<36, 256, 0, stream>>>(ln_w, ENC, 3, WP);
    conv_wmma_k<<<8192, 64, 0, stream>>>(Xb, ENC, WP, ln_b, 3, 1, RES, 0, 1); // relu

    ln_reduce_k<<<BB * 256, 256, 0, stream>>>(ZN, RES, PART);
    ln_stats_k<<<BB, 256, 0, stream>>>(PART, STATS);
    ln_apply_k<<<2048, 256, 0, stream>>>(ZN, RES, n2_w, n2_b, STATS, X);
  }
  final_conv_k<<<512, 256, 0, stream>>>(X, (const float*)d_in[58],
                                        (const float*)d_in[59], (float*)d_out);
}
